// MOE_PositionwiseFeedForward_2980707304165
// MI455X (gfx1250) — matrix operand order repaired and checked by reading, not compiled, not run on hardware
//
#include <hip/hip_runtime.h>

// Problem dims (B=8, S=1024 -> 8192 tokens), D=1024, H=4096, E=3, R=8
#define D_    1024
#define H_    4096
#define E_    3
#define R_    8
#define NTOK  8192
#define K1    1056   // D + 32 (E*R = 24, padded to 32 for K%32==0)
#define K2    4128   // H + 32
#define SCALING 0.125f

typedef float v8f   __attribute__((ext_vector_type(8)));
typedef __bf16 v16bf __attribute__((ext_vector_type(16)));
typedef unsigned int u32;
typedef u32 u32x4 __attribute__((ext_vector_type(4)));

union Frag { u32x4 h[2]; v16bf v; };

__device__ inline __bf16 f2bf(float f) {
  u32 u = __builtin_bit_cast(u32, f);
  u += 0x7fffu + ((u >> 16) & 1u);              // round-to-nearest-even
  unsigned short s = (unsigned short)(u >> 16);
  return __builtin_bit_cast(__bf16, s);
}
__device__ inline float bf2f(__bf16 h) {
  unsigned short s = __builtin_bit_cast(unsigned short, h);
  u32 u = ((u32)s) << 16;
  return __builtin_bit_cast(float, u);
}

// ---------------------------------------------------------------------------
// Weight prep: build bf16 transposed augmented panels  Wt[n*K + k]
//   k <  Din       : W[k, n]
//   k <  Din + 24  : Blora[(k-Din), n]   (E*R rows concatenated)
//   else           : 0 (pad)
// ---------------------------------------------------------------------------
__global__ void prep_w1_kernel(const float* __restrict__ W1,
                               const float* __restrict__ B1,
                               __bf16* __restrict__ Wt) {
  const int idx = blockIdx.x * blockDim.x + threadIdx.x;
  if (idx >= H_ * K1) return;
  const int n = idx / K1, k = idx - n * K1;
  float v = 0.f;
  if (k < D_)            v = W1[(size_t)k * H_ + n];
  else if (k < D_ + 24)  v = B1[(size_t)(k - D_) * H_ + n];
  Wt[idx] = f2bf(v);
}

__global__ void prep_w2_kernel(const float* __restrict__ W2,
                               const float* __restrict__ B2,
                               __bf16* __restrict__ Wt) {
  const int idx = blockIdx.x * blockDim.x + threadIdx.x;
  if (idx >= D_ * K2) return;
  const int n = idx / K2, k = idx - n * K2;
  float v = 0.f;
  if (k < H_)            v = W2[(size_t)k * D_ + n];
  else if (k < H_ + 24)  v = B2[(size_t)(k - H_) * D_ + n];
  Wt[idx] = f2bf(v);
}

// ---------------------------------------------------------------------------
// Router 1: per token, copy x -> bf16 X_aug[:,0:1024] and compute tail
//   tail[e*8+r] = SCALING * softmax(x@R1)[e] * (x@A1[e])[r]
// One wave per token; x row staged in LDS; 27 dot products across lanes.
// ---------------------------------------------------------------------------
__global__ __launch_bounds__(256, 1)
void router1_kernel(const float* __restrict__ x, const float* __restrict__ A1,
                    const float* __restrict__ R1, __bf16* __restrict__ Xa) {
  __shared__ float xrow[8][D_];
  __shared__ float sc[8][32];
  const int lane = threadIdx.x & 31;
  const int wave = threadIdx.x >> 5;
  const int t = blockIdx.x * 8 + wave;
  const float* xr = x + (size_t)t * D_;
  __bf16* xa = Xa + (size_t)t * K1;

  for (int j = lane; j < D_; j += 32) {
    float v = xr[j];
    xrow[wave][j] = v;
    xa[j] = f2bf(v);
  }
  __syncthreads();

  float acc = 0.f;
  if (lane < 24) {                              // z_{e,r} = x . A1[e,:,r]
    const int e = lane >> 3, r = lane & 7;
    const float* ap = A1 + (size_t)e * D_ * R_ + r;
    #pragma unroll 4
    for (int d = 0; d < D_; ++d) acc += xrow[wave][d] * ap[(size_t)d * R_];
  } else if (lane < 27) {                       // logits = x . R1[:,e]
    const float* rp = R1 + (lane - 24);
    #pragma unroll 4
    for (int d = 0; d < D_; ++d) acc += xrow[wave][d] * rp[(size_t)d * E_];
  }
  sc[wave][lane] = acc;
  __syncthreads();

  const float g0 = sc[wave][24], g1 = sc[wave][25], g2 = sc[wave][26];
  const float gm = fmaxf(g0, fmaxf(g1, g2));
  const float e0 = __expf(g0 - gm), e1 = __expf(g1 - gm), e2 = __expf(g2 - gm);
  const float inv = 1.0f / (e0 + e1 + e2);
  float val = 0.f;
  if (lane < 24) {
    const int e = lane >> 3;
    const float ge = (e == 0 ? e0 : (e == 1 ? e1 : e2)) * inv;
    val = SCALING * ge * sc[wave][lane];
  }
  xa[D_ + lane] = f2bf(val);                    // lanes 24..31 write the zero pad
}

// ---------------------------------------------------------------------------
// Router 2: same, reading the bf16 hidden row (Ha[:,0:4096]), fills Ha tail.
// Row is staged in LDS (packed bf16 pairs).
// ---------------------------------------------------------------------------
__global__ __launch_bounds__(256, 1)
void router2_kernel(__bf16* __restrict__ Ha, const float* __restrict__ A2,
                    const float* __restrict__ R2) {
  __shared__ float hrow[8][H_ / 2];   // staged as bf16 pairs packed in u32
  __shared__ float sc[8][32];
  const int lane = threadIdx.x & 31;
  const int wave = threadIdx.x >> 5;
  const int t = blockIdx.x * 8 + wave;
  const u32* hr = (const u32*)(Ha + (size_t)t * K2);   // 2 bf16 per u32
  u32* hs = (u32*)hrow[wave];

  for (int j = lane; j < H_ / 2; j += 32) hs[j] = hr[j];
  __syncthreads();

  float acc = 0.f;
  if (lane < 24) {
    const int e = lane >> 3, r = lane & 7;
    const float* ap = A2 + (size_t)e * H_ * R_ + r;
    for (int d2 = 0; d2 < H_ / 2; ++d2) {
      const u32 pk = hs[d2];
      const float h0 = __builtin_bit_cast(float, pk << 16);
      const float h1 = __builtin_bit_cast(float, pk & 0xffff0000u);
      acc += h0 * ap[(size_t)(2 * d2) * R_] + h1 * ap[(size_t)(2 * d2 + 1) * R_];
    }
  } else if (lane < 27) {
    const float* rp = R2 + (lane - 24);
    for (int d2 = 0; d2 < H_ / 2; ++d2) {
      const u32 pk = hs[d2];
      const float h0 = __builtin_bit_cast(float, pk << 16);
      const float h1 = __builtin_bit_cast(float, pk & 0xffff0000u);
      acc += h0 * rp[(size_t)(2 * d2) * E_] + h1 * rp[(size_t)(2 * d2 + 1) * E_];
    }
  }
  sc[wave][lane] = acc;
  __syncthreads();

  const float g0 = sc[wave][24], g1 = sc[wave][25], g2 = sc[wave][26];
  const float gm = fmaxf(g0, fmaxf(g1, g2));
  const float e0 = __expf(g0 - gm), e1 = __expf(g1 - gm), e2 = __expf(g2 - gm);
  const float inv = 1.0f / (e0 + e1 + e2);
  float val = 0.f;
  if (lane < 24) {
    const int e = lane >> 3;
    const float ge = (e == 0 ? e0 : (e == 1 ? e1 : e2)) * inv;
    val = SCALING * ge * sc[wave][lane];
  }
  Ha[(size_t)t * K2 + H_ + lane] = f2bf(val);
}

// ---------------------------------------------------------------------------
// WMMA GEMM: C[M,N] = A[M,K](bf16,row-major) @ Bt[N,K](bf16,row-major)^T + bias
// Block = 256 thr = 8 waves -> 128x256 tile; each wave does 64x64 = 4x4 WMMA.
// Addressing: per-lane u32 BYTE offsets on an i8 base pointer; the K-advance
// is folded into the uniform base so loads can use the GVS saddr+voffset form
// (SGPR_U64 + VGPR_I32), keeping per-iteration address VALU near zero.
// ---------------------------------------------------------------------------
template <bool RELU_BF16OUT>
__global__ __launch_bounds__(256, 1)
void gemm_bf16_wmma(const __bf16* __restrict__ A, const __bf16* __restrict__ Bt,
                    const float* __restrict__ bias, void* __restrict__ Cout,
                    int K, int ldc) {
  const int lane  = threadIdx.x & 31;
  const int wave  = threadIdx.x >> 5;
  const int l16   = lane & 15;
  const int lhalf = lane >> 4;
  const int M0 = blockIdx.y * 128 + (wave >> 2) * 64;
  const int N0 = blockIdx.x * 256 + (wave & 3) * 64;

  v8f acc[4][4] = {};

  // Loop-invariant per-lane BYTE offsets (< 2^27, fits signed 32-bit).
  u32 aOffB[4], bOffB[4];
  #pragma unroll
  for (int mt = 0; mt < 4; ++mt)
    aOffB[mt] = ((u32)(M0 + mt * 16 + l16) * (u32)K + (u32)(lhalf * 8)) * 2u;
  #pragma unroll
  for (int nt = 0; nt < 4; ++nt)
    bOffB[nt] = ((u32)(N0 + nt * 16 + l16) * (u32)K + (u32)(lhalf * 8)) * 2u;

  const char* Abase = (const char*)A;
  const char* Bbase = (const char*)Bt;

  for (int k = 0; k < K; k += 32) {
    // Uniform (scalar) K-advance: base pointers step by 64 bytes.
    const char* Ak = Abase + (size_t)((u32)k * 2u);
    const char* Bk = Bbase + (size_t)((u32)k * 2u);

    Frag af[4], bf[4];
    #pragma unroll
    for (int mt = 0; mt < 4; ++mt) {
      af[mt].h[0] = *(const u32x4*)(Ak + aOffB[mt]);
      af[mt].h[1] = *(const u32x4*)(Ak + aOffB[mt] + 32u);
    }
    #pragma unroll
    for (int nt = 0; nt < 4; ++nt) {
      bf[nt].h[0] = *(const u32x4*)(Bk + bOffB[nt]);
      bf[nt].h[1] = *(const u32x4*)(Bk + bOffB[nt] + 32u);
    }
    // Unconditional speculative prefetch of the K-chunk two ahead (k+64) (dropped if OOB).
    __builtin_prefetch(Ak + aOffB[0] + 128u, 0, 0);
    __builtin_prefetch(Bk + bOffB[0] + 128u, 0, 0);

    #pragma unroll
    for (int mt = 0; mt < 4; ++mt)
      #pragma unroll
      for (int nt = 0; nt < 4; ++nt)
        acc[mt][nt] = __builtin_amdgcn_wmma_f32_16x16x32_bf16(
            false, af[mt].v, false, bf[nt].v, (short)0, acc[mt][nt],
            false, false);
  }

  // Epilogue. C layout: VGPR i holds (m = i + lhalf*8, n = l16) of each tile.
  #pragma unroll
  for (int mt = 0; mt < 4; ++mt) {
    #pragma unroll
    for (int nt = 0; nt < 4; ++nt) {
      const int col  = N0 + nt * 16 + l16;
      const int row0 = M0 + mt * 16 + lhalf * 8;
      const float bv = bias[col];
      #pragma unroll
      for (int i = 0; i < 8; ++i) {
        float v = acc[mt][nt][i] + bv;
        const u32 off = (u32)(row0 + i) * (u32)ldc + (u32)col;
        if (RELU_BF16OUT) {
          v = fmaxf(v, 0.0f);
          ((__bf16*)Cout)[off] = f2bf(v);
        } else {
          ((float*)Cout)[off] = v;
        }
      }
    }
  }
}

// ---------------------------------------------------------------------------
extern "C" void kernel_launch(void* const* d_in, const int* in_sizes, int n_in,
                              void* d_out, int out_size, void* d_ws, size_t ws_size,
                              hipStream_t stream) {
  (void)in_sizes; (void)n_in; (void)out_size; (void)ws_size;
  const float* x  = (const float*)d_in[0];
  const float* W1 = (const float*)d_in[1];
  const float* b1 = (const float*)d_in[2];
  const float* A1 = (const float*)d_in[3];
  const float* B1 = (const float*)d_in[4];
  const float* R1 = (const float*)d_in[5];
  const float* W2 = (const float*)d_in[6];
  const float* b2 = (const float*)d_in[7];
  const float* A2 = (const float*)d_in[8];
  const float* B2 = (const float*)d_in[9];
  const float* R2 = (const float*)d_in[10];

  char* ws = (char*)d_ws;
  size_t off = 0;
  auto take = [&](size_t bytes) -> char* {
    char* p = ws + off;
    off = (off + bytes + 255) & ~(size_t)255;
    return p;
  };
  __bf16* Wt1 = (__bf16*)take((size_t)H_ * K1 * sizeof(__bf16));   //  8.7 MB
  __bf16* Wt2 = (__bf16*)take((size_t)D_ * K2 * sizeof(__bf16));   //  8.5 MB
  __bf16* Xa  = (__bf16*)take((size_t)NTOK * K1 * sizeof(__bf16)); // 17.3 MB
  __bf16* Ha  = (__bf16*)take((size_t)NTOK * K2 * sizeof(__bf16)); // 67.6 MB

  prep_w1_kernel<<<(H_ * K1 + 255) / 256, 256, 0, stream>>>(W1, B1, Wt1);
  prep_w2_kernel<<<(D_ * K2 + 255) / 256, 256, 0, stream>>>(W2, B2, Wt2);

  router1_kernel<<<NTOK / 8, 256, 0, stream>>>(x, A1, R1, Xa);

  // h = relu(X_aug @ Wt1^T + b1) -> bf16, row stride K2 (tail filled by router2)
  gemm_bf16_wmma<true><<<dim3(H_ / 256, NTOK / 128), 256, 0, stream>>>(
      Xa, Wt1, b1, (void*)Ha, K1, K2);

  router2_kernel<<<NTOK / 8, 256, 0, stream>>>(Ha, A2, R2);

  // out = H_aug @ Wt2^T + b2 -> fp32
  gemm_bf16_wmma<false><<<dim3(D_ / 256, NTOK / 128), 256, 0, stream>>>(
      Ha, Wt2, b2, d_out, K2, D_);
}
